// RealAgnosticResidualInteractionLayer_10144712753178
// MI455X (gfx1250) — matrix operand order repaired and checked by reading, not compiled, not run on hardware
//
#include <hip/hip_runtime.h>
#include <hip/hip_bf16.h>

typedef __attribute__((ext_vector_type(16))) __bf16 v16bf;
typedef __attribute__((ext_vector_type(8)))  float  v8f;

#define N_NODES 50000
#define N_EDGES 600000
#define RS_CONST 1.6765324703310907f

// workspace layout (float units)
#define WS_X0   0L           // N*32
#define WS_X1   1600000L     // N*96
#define WS_MSG0 6400000L     // N*64   (paths p0,p3)
#define WS_MSG1 9600000L     // N*288  (paths p1,p2,p4 x 3 comps)
#define WS_WPK  24000000L    // 10240 uint32 of packed bf16 B-fragments (40 KB)

__device__ __forceinline__ float rsilu(float x) {
    // RS * x * sigmoid(x); fast v_rcp_f32 instead of IEEE divide --
    // activations are bf16-precision anyway.
    return RS_CONST * x * __builtin_amdgcn_rcpf(1.f + __expf(-x));
}

// ---------------------------------------------------------------------------
// Pack MLP weights into bf16 WMMA B-fragment order (scales folded in).
// Fragment f: 32 lanes x 8 dwords. Lane l holds column n=l%16, VGPR v holds
// K pair k = 2(v%4) + 8*(l/16) + 16*(v/4) (+kc*32 chunk offset).
// Frags: [0..3]=L0 (K pad 8->32), [4..11]=L1, [12..19]=L2, [20..39]=L3.
// ---------------------------------------------------------------------------
__global__ __launch_bounds__(256) void prep_weights(
    const float* __restrict__ Wr0, const float* __restrict__ Wr1,
    const float* __restrict__ Wr2, const float* __restrict__ Wr3,
    unsigned* __restrict__ wpack) {
    int word = blockIdx.x * 256 + threadIdx.x;   // 40*256 words
    int f = word >> 8;
    int l = (word >> 3) & 31;
    int v = word & 7;
    int nl = l & 15;
    int klocal = (v & 3) * 2 + 8 * (l >> 4) + (v >> 2) * 16;
    float v0 = 0.f, v1 = 0.f;
    if (f < 4) {
        int n = f * 16 + nl;
        const float s = 0.35355339059327373f;     // 1/sqrt(8)
        if (klocal < 8)     v0 = Wr0[n * 8 + klocal] * s;
        if (klocal + 1 < 8) v1 = Wr0[n * 8 + klocal + 1] * s;
    } else if (f < 12) {
        int q = f - 4; int n = (q >> 1) * 16 + nl; int k = (q & 1) * 32 + klocal;
        v0 = Wr1[n * 64 + k] * 0.125f; v1 = Wr1[n * 64 + k + 1] * 0.125f;
    } else if (f < 20) {
        int q = f - 12; int n = (q >> 1) * 16 + nl; int k = (q & 1) * 32 + klocal;
        v0 = Wr2[n * 64 + k] * 0.125f; v1 = Wr2[n * 64 + k + 1] * 0.125f;
    } else {
        int q = f - 20; int n = (q >> 1) * 16 + nl; int k = (q & 1) * 32 + klocal;
        v0 = Wr3[n * 64 + k] * 0.125f; v1 = Wr3[n * 64 + k + 1] * 0.125f;
    }
    unsigned short lo = __builtin_bit_cast(unsigned short, (__bf16)v0);
    unsigned short hi = __builtin_bit_cast(unsigned short, (__bf16)v1);
    wpack[word] = (unsigned)lo | ((unsigned)hi << 16);
}

__global__ __launch_bounds__(256) void zero_kernel(float4* __restrict__ p, long n4) {
    long i = (long)blockIdx.x * blockDim.x + threadIdx.x;
    if (i < n4) p[i] = make_float4(0.f, 0.f, 0.f, 0.f);
}

// ---------------------------------------------------------------------------
// Node-side: sc output + x0/x1 precompute. Thread = (node, out-channel d).
// ---------------------------------------------------------------------------
__global__ __launch_bounds__(256) void node_kernel(
    const float* __restrict__ node_attrs, const float* __restrict__ node_feats,
    const float* __restrict__ W_first, const float* __restrict__ W_skip,
    float* __restrict__ x0, float* __restrict__ x1, float* __restrict__ sc_out) {
    long gid = (long)blockIdx.x * 256 + threadIdx.x;
    if (gid >= (long)N_NODES * 32) return;
    int n = (int)(gid >> 5);
    int d = (int)(gid & 31);
    float attrs[10];
#pragma unroll
    for (int v = 0; v < 10; ++v) attrs[v] = node_attrs[(long)n * 10 + v];
    float sc0 = 0.f, sc1x = 0.f, sc1y = 0.f, sc1z = 0.f;
    float xx0 = 0.f, xx1x = 0.f, xx1y = 0.f, xx1z = 0.f;
    for (int cc = 0; cc < 32; ++cc) {
        const float* fp = &node_feats[((long)n * 32 + cc) * 4];
        float f0 = fp[0], f1 = fp[1], f2 = fp[2], f3 = fp[3];
        float G0 = 0.f, G1 = 0.f;
#pragma unroll
        for (int v = 0; v < 10; ++v) {
            G0 += attrs[v] * W_skip[((long)(cc) * 10 + v) * 32 + d];
            G1 += attrs[v] * W_skip[((long)(32 + cc) * 10 + v) * 32 + d];
        }
        sc0 += f0 * G0;
        sc1x += f1 * G1; sc1y += f2 * G1; sc1z += f3 * G1;
        float wf0 = W_first[d * 32 + cc];
        float wf1 = W_first[1024 + d * 32 + cc];
        xx0 += f0 * wf0;
        xx1x += f1 * wf1; xx1y += f2 * wf1; xx1z += f3 * wf1;
    }
    const float ssc = 0.05590169943749474f;   // 1/sqrt(320)
    const float sx  = 0.17677669529663687f;   // 1/sqrt(32)
    float* scp = &sc_out[((long)n * 32 + d) * 4];
    scp[0] = sc0 * ssc; scp[1] = sc1x * ssc; scp[2] = sc1y * ssc; scp[3] = sc1z * ssc;
    x0[(long)n * 32 + d] = xx0 * sx;
    float* xp = &x1[((long)n * 32 + d) * 3];
    xp[0] = xx1x * sx; xp[1] = xx1y * sx; xp[2] = xx1z * sx;
}

// ---------------------------------------------------------------------------
// Fused edge kernel: one wave per 16-edge tile.
//   bf16 WMMA MLP 8->64->64->64->160 (LDS ping-pong between layers),
//   then per-edge tensor-product messages with register accumulation over
//   the sorted idx_i run, flushed via global_atomic_add_f32.
// ---------------------------------------------------------------------------
__global__ __launch_bounds__(32) void edge_kernel(
    const float* __restrict__ edge_feats, const float* __restrict__ edge_attrs,
    const int* __restrict__ idx_i, const int* __restrict__ idx_j,
    const float* __restrict__ x0, const float* __restrict__ x1,
    const unsigned* __restrict__ wpack,
    float* __restrict__ msg0, float* __restrict__ msg1) {
    __shared__ float bufA[2][16][64];
    __shared__ float bufW[16][160];
    const int lane  = threadIdx.x;
    const int ml    = lane & 15;
    const int khalf = lane >> 4;
    const long tileBase = (long)blockIdx.x * 16;
    const v16bf* bfrags = reinterpret_cast<const v16bf*>(wpack);

    // ----- layer 0: (16x32, K padded from 8) x (32x64) -----
    v16bf a0;
    {
        long e = tileBase + ml;
        if (e >= N_EDGES) e = N_EDGES - 1;
#pragma unroll
        for (int v = 0; v < 8; ++v) {
            int k = (v & 3) * 2 + khalf * 8 + (v >> 2) * 16;
            float f0 = (k < 8)     ? edge_feats[e * 8 + k]     : 0.f;
            float f1 = (k + 1 < 8) ? edge_feats[e * 8 + k + 1] : 0.f;
            a0[2 * v]     = (__bf16)f0;
            a0[2 * v + 1] = (__bf16)f1;
        }
    }
#pragma unroll
    for (int t = 0; t < 4; ++t) {
        v8f c = {};
        v16bf b = bfrags[t * 32 + lane];
        c = __builtin_amdgcn_wmma_f32_16x16x32_bf16(false, a0, false, b, (short)0, c, false, false);
#pragma unroll
        for (int r = 0; r < 8; ++r)
            bufA[0][r + 8 * khalf][t * 16 + ml] = rsilu(c[r]);
    }
    __syncthreads();

    // ----- layers 1 & 2: (16x64) x (64x64), ping-pong LDS -----
#pragma unroll
    for (int L = 0; L < 2; ++L) {
        const int src = L, dst = 1 - L;
        const int fbase = 4 + L * 8;
        v16bf alo, ahi;
#pragma unroll
        for (int v = 0; v < 8; ++v) {
            int k = (v & 3) * 2 + khalf * 8 + (v >> 2) * 16;
            alo[2 * v]     = (__bf16)bufA[src][ml][k];
            alo[2 * v + 1] = (__bf16)bufA[src][ml][k + 1];
            ahi[2 * v]     = (__bf16)bufA[src][ml][32 + k];
            ahi[2 * v + 1] = (__bf16)bufA[src][ml][32 + k + 1];
        }
#pragma unroll
        for (int t = 0; t < 4; ++t) {
            v8f c = {};
            v16bf b0 = bfrags[(fbase + t * 2 + 0) * 32 + lane];
            v16bf b1 = bfrags[(fbase + t * 2 + 1) * 32 + lane];
            c = __builtin_amdgcn_wmma_f32_16x16x32_bf16(false, alo, false, b0, (short)0, c, false, false);
            c = __builtin_amdgcn_wmma_f32_16x16x32_bf16(false, ahi, false, b1, (short)0, c, false, false);
#pragma unroll
            for (int r = 0; r < 8; ++r)
                bufA[dst][r + 8 * khalf][t * 16 + ml] = rsilu(c[r]);
        }
        __syncthreads();
    }

    // ----- layer 3: (16x64) x (64x160) -> w, no activation -----
    {
        v16bf alo, ahi;
#pragma unroll
        for (int v = 0; v < 8; ++v) {
            int k = (v & 3) * 2 + khalf * 8 + (v >> 2) * 16;
            alo[2 * v]     = (__bf16)bufA[0][ml][k];
            alo[2 * v + 1] = (__bf16)bufA[0][ml][k + 1];
            ahi[2 * v]     = (__bf16)bufA[0][ml][32 + k];
            ahi[2 * v + 1] = (__bf16)bufA[0][ml][32 + k + 1];
        }
#pragma unroll
        for (int t = 0; t < 10; ++t) {
            v8f c = {};
            v16bf b0 = bfrags[(20 + t * 2 + 0) * 32 + lane];
            v16bf b1 = bfrags[(20 + t * 2 + 1) * 32 + lane];
            c = __builtin_amdgcn_wmma_f32_16x16x32_bf16(false, alo, false, b0, (short)0, c, false, false);
            c = __builtin_amdgcn_wmma_f32_16x16x32_bf16(false, ahi, false, b1, (short)0, c, false, false);
#pragma unroll
            for (int r = 0; r < 8; ++r)
                bufW[r + 8 * khalf][t * 16 + ml] = c[r];
        }
    }
    __syncthreads();

    // ----- message phase: lane = channel c, loop edges in tile -----
    const int c_ch = lane;
    float acc0 = 0.f, acc3 = 0.f;
    float acc1[3] = {0.f, 0.f, 0.f}, acc2[3] = {0.f, 0.f, 0.f}, acc4[3] = {0.f, 0.f, 0.f};
    int cur_i = -1;
    auto flush = [&](int node) {
        atomicAdd(&msg0[(long)node * 64 + c_ch], acc0);
        atomicAdd(&msg0[(long)node * 64 + 32 + c_ch], acc3);
#pragma unroll
        for (int mm = 0; mm < 3; ++mm) {
            atomicAdd(&msg1[((long)node * 96 + c_ch) * 3 + mm], acc1[mm]);
            atomicAdd(&msg1[((long)node * 96 + 32 + c_ch) * 3 + mm], acc2[mm]);
            atomicAdd(&msg1[((long)node * 96 + 64 + c_ch) * 3 + mm], acc4[mm]);
        }
        acc0 = acc3 = 0.f;
#pragma unroll
        for (int mm = 0; mm < 3; ++mm) { acc1[mm] = 0.f; acc2[mm] = 0.f; acc4[mm] = 0.f; }
    };
    for (int e = 0; e < 16; ++e) {
        long ge = tileBase + e;
        if (ge >= N_EDGES) break;                 // uniform across the wave
        int i = idx_i[ge];
        if (i != cur_i) {
            if (cur_i >= 0) flush(cur_i);         // sorted idx_i: ~12x fewer atomics
            cur_i = i;
        }
        int j = idx_j[ge];
        float y0 = edge_attrs[ge * 4 + 0];
        float yx = edge_attrs[ge * 4 + 1];
        float yy = edge_attrs[ge * 4 + 2];
        float yz = edge_attrs[ge * 4 + 3];
        float xj0v = x0[(long)j * 32 + c_ch];
        const float* xp = &x1[((long)j * 32 + c_ch) * 3];
        float ax = xp[0], ay = xp[1], az = xp[2];
        float w0 = bufW[e][c_ch];
        float w1 = bufW[e][32 + c_ch];
        float w2 = bufW[e][64 + c_ch];
        float w3 = bufW[e][96 + c_ch];
        float w4 = bufW[e][128 + c_ch];
        acc0 += w0 * xj0v * y0;
        acc3 += w3 * (ax * yx + ay * yy + az * yz);
        float t1 = w1 * xj0v;
        acc1[0] += t1 * yx; acc1[1] += t1 * yy; acc1[2] += t1 * yz;
        float t2 = w2 * y0;
        acc2[0] += t2 * ax; acc2[1] += t2 * ay; acc2[2] += t2 * az;
        acc4[0] += w4 * (ay * yz - az * yy);
        acc4[1] += w4 * (az * yx - ax * yz);
        acc4[2] += w4 * (ax * yy - ay * yx);
    }
    if (cur_i >= 0) flush(cur_i);
}

// ---------------------------------------------------------------------------
// Final node contraction with W_second -> message output.
// ---------------------------------------------------------------------------
__global__ __launch_bounds__(256) void out_kernel(
    const float* __restrict__ msg0, const float* __restrict__ msg1,
    const float* __restrict__ W_second0, const float* __restrict__ W_second1,
    float* __restrict__ message) {
    long gid = (long)blockIdx.x * 256 + threadIdx.x;
    if (gid >= (long)N_NODES * 32) return;
    int n = (int)(gid >> 5);
    int d = (int)(gid & 31);
    float o0 = 0.f;
    for (int pc = 0; pc < 64; ++pc)
        o0 += msg0[(long)n * 64 + pc] * W_second0[d * 64 + pc];
    float o1x = 0.f, o1y = 0.f, o1z = 0.f;
    for (int pc = 0; pc < 96; ++pc) {
        float wv = W_second1[d * 96 + pc];
        const float* mp = &msg1[((long)n * 96 + pc) * 3];
        o1x += mp[0] * wv; o1y += mp[1] * wv; o1z += mp[2] * wv;
    }
    const float s0 = 1.f / (8.f * 12.f);                    // 1/(sqrt(64)*AVG)
    const float s1 = 1.f / (9.797958971132712f * 12.f);     // 1/(sqrt(96)*AVG)
    float* op = &message[((long)n * 32 + d) * 4];
    op[0] = o0 * s0;
    op[1] = o1x * s1; op[2] = o1y * s1; op[3] = o1z * s1;
}

extern "C" void kernel_launch(void* const* d_in, const int* in_sizes, int n_in,
                              void* d_out, int out_size, void* d_ws, size_t ws_size,
                              hipStream_t stream) {
    (void)in_sizes; (void)n_in; (void)out_size; (void)ws_size;
    const float* node_attrs = (const float*)d_in[0];
    const float* node_feats = (const float*)d_in[1];
    const float* edge_attrs = (const float*)d_in[2];
    const float* edge_feats = (const float*)d_in[3];
    const float* W_first    = (const float*)d_in[4];
    const float* Wr0        = (const float*)d_in[5];
    const float* Wr1        = (const float*)d_in[6];
    const float* Wr2        = (const float*)d_in[7];
    const float* Wr3        = (const float*)d_in[8];
    const float* W_second0  = (const float*)d_in[9];
    const float* W_second1  = (const float*)d_in[10];
    const float* W_skip     = (const float*)d_in[11];
    const int*   idx_i      = (const int*)d_in[12];
    const int*   idx_j      = (const int*)d_in[13];

    float* ws    = (float*)d_ws;
    float* x0    = ws + WS_X0;
    float* x1    = ws + WS_X1;
    float* msg0  = ws + WS_MSG0;
    float* msg1  = ws + WS_MSG1;
    unsigned* wpack = (unsigned*)(ws + WS_WPK);

    float* message = (float*)d_out;
    float* sc_out  = (float*)d_out + (long)N_NODES * 32 * 4;

    // 1. pack MLP weights into WMMA B-fragment layout (bf16, scales folded)
    prep_weights<<<40, 256, 0, stream>>>(Wr0, Wr1, Wr2, Wr3, wpack);

    // 2. zero message accumulators (msg0|msg1 contiguous: N*352 floats)
    long nz4 = ((long)N_NODES * 352) / 4;
    zero_kernel<<<(int)((nz4 + 255) / 256), 256, 0, stream>>>((float4*)msg0, nz4);

    // 3. node-side skip connection + x0/x1 precompute
    node_kernel<<<(N_NODES * 32 + 255) / 256, 256, 0, stream>>>(
        node_attrs, node_feats, W_first, W_skip, x0, x1, sc_out);

    // 4. fused edge MLP (WMMA) + tensor-product messages + atomic scatter
    edge_kernel<<<N_EDGES / 16, 32, 0, stream>>>(
        edge_feats, edge_attrs, idx_i, idx_j, x0, x1, wpack, msg0, msg1);

    // 5. final contraction with W_second -> message output
    out_kernel<<<(N_NODES * 32 + 255) / 256, 256, 0, stream>>>(
        msg0, msg1, W_second0, W_second1, message);
}
